// Unet_53644141527392
// MI455X (gfx1250) — matrix operand order repaired and checked by reading, not compiled, not run on hardware
//
#include <hip/hip_runtime.h>
#include <hip/hip_bf16.h>

typedef __bf16 bf16_t;
typedef __attribute__((ext_vector_type(16))) __bf16 v16bf;
typedef __attribute__((ext_vector_type(8)))  __bf16 v8bf;
typedef __attribute__((ext_vector_type(8)))  float  v8f;

#if defined(__gfx1250__) && __has_builtin(__builtin_amdgcn_tensor_load_to_lds)
#define HAVE_TDM 1
#else
#define HAVE_TDM 0
#endif
#if defined(__gfx1250__) && __has_builtin(__builtin_amdgcn_global_load_async_to_lds_b128)
#define HAVE_ASYNC_COPY 1
#else
#define HAVE_ASYNC_COPY 0
#endif

#if defined(__gfx1250__)
typedef __attribute__((address_space(3))) char lds_char_t;
typedef __attribute__((ext_vector_type(4))) int v4i;
typedef __attribute__((address_space(1))) v4i g_v4i;   // global int4
typedef __attribute__((address_space(3))) v4i l_v4i;   // LDS int4
#endif

#if HAVE_TDM
typedef __attribute__((ext_vector_type(4))) unsigned int u32x4;
typedef __attribute__((ext_vector_type(8))) int i32x8;
typedef __attribute__((ext_vector_type(4))) int i32x4;

// 1-D TDM copy: n8 elements of 8 bytes, global -> LDS.  Descriptor per ISA ch8:
// group0: count=1 | lds_addr | global_addr[56:0] | type=2
// group1: data_size=8B, tensor_dim0=n8, tensor_dim1=1, tile_dim0=n8 (1-D tile)
__device__ __forceinline__ void tdm_copy_1d(const bf16_t* gsrc, unsigned lds_byte_off,
                                            unsigned n8) {
  unsigned long long ga = (unsigned long long)(const void*)gsrc;
  u32x4 g0;
  g0[0] = 1u;                                            // count=1, user mode
  g0[1] = lds_byte_off;                                  // LDS byte address
  g0[2] = (unsigned)ga;                                  // global_addr[31:0]
  g0[3] = (unsigned)((ga >> 32) & 0x01ffffffu) | (2u << 30); // [56:32] | type=2
  i32x8 g1;
  g1[0] = (int)(3u << 16);                               // data_size = 8 bytes
  g1[1] = (int)((n8 & 0xffffu) << 16);                   // tensor_dim0[15:0]
  g1[2] = (int)(((n8 >> 16) & 0xffffu) | (1u << 16));    // td0[31:16], tensor_dim1=1
  g1[3] = (int)((n8 & 0xffffu) << 16);                   // tile_dim0 = n8
  g1[4] = 0;                                             // tile_dim1/2 unused
  g1[5] = (int)n8;                                       // tensor_dim0_stride
  g1[6] = 0;
  g1[7] = 0;
  i32x4 gz4 = {0, 0, 0, 0};
  i32x8 gz8 = {0, 0, 0, 0, 0, 0, 0, 0};
  __builtin_amdgcn_tensor_load_to_lds(g0, g1, gz4, gz4, gz8, 0);
}
#endif

// ---------------------------------------------------------------------------
// Hypernet: (B,2) -> leaky -> (B,8) -> leaky -> (B,32) -> (B,32)
// ---------------------------------------------------------------------------
__global__ void hypernet_kernel(const float* __restrict__ hp,
                                const float* __restrict__ W1, const float* __restrict__ b1,
                                const float* __restrict__ W2, const float* __restrict__ b2,
                                const float* __restrict__ Wo, const float* __restrict__ bo,
                                float* __restrict__ hyp_out /* 4x32 */) {
  __shared__ float h1[4 * 8];
  __shared__ float h2[4 * 32];
  int t = threadIdx.x;
  if (t < 32) {
    int b = t >> 3, j = t & 7;
    float a = b1[j];
    for (int k = 0; k < 2; ++k) a += hp[b * 2 + k] * W1[k * 8 + j];
    h1[t] = a >= 0.f ? a : 0.01f * a;
  }
  __syncthreads();
  if (t < 128) {
    int b = t >> 5, j = t & 31;
    float a = b2[j];
    for (int k = 0; k < 8; ++k) a += h1[b * 8 + k] * W2[k * 32 + j];
    h2[t] = a >= 0.f ? a : 0.01f * a;
  }
  __syncthreads();
  if (t < 128) {
    int b = t >> 5, j = t & 31;
    float a = bo[j];
    for (int k = 0; k < 32; ++k) a += h2[b * 32 + k] * Wo[k * 32 + j];
    hyp_out[t] = a;
  }
}

// ---------------------------------------------------------------------------
// Per-sample weight generation: out layout [B][ks*ks][cout][cinp] bf16,
// channel-padded to cinp (zeros beyond actual cin).
// ---------------------------------------------------------------------------
__global__ void genw_kernel(const float* __restrict__ hyp,
                            const float* __restrict__ Wk, const float* __restrict__ bk,
                            bf16_t* __restrict__ wout,
                            int cin, int cinp, int cout, int ks, int total) {
  int idx = blockIdx.x * blockDim.x + threadIdx.x;
  if (idx >= total) return;
  int ci  = idx % cinp;
  int co  = (idx / cinp) % cout;
  int tap = (idx / (cinp * cout)) % (ks * ks);
  int b   = idx / (cinp * cout * ks * ks);
  float v = 0.f;
  if (ci < cin) {
    int dy = tap / ks, dx = tap % ks;
    int j = ((co * cin + ci) * ks + dy) * ks + dx;
    int ncol = cout * cin * ks * ks;
    v = bk[j];
    for (int u = 0; u < 32; ++u) v += hyp[b * 32 + u] * Wk[u * ncol + j];
  }
  wout[idx] = (bf16_t)v;
}

__global__ void genb_kernel(const float* __restrict__ hyp,
                            const float* __restrict__ Wb, const float* __restrict__ bb,
                            float* __restrict__ bout, int cout) {
  int t = blockIdx.x * blockDim.x + threadIdx.x;
  if (t >= 4 * cout) return;
  int co = t % cout, b = t / cout;
  float v = bb[co];
  for (int u = 0; u < 32; ++u) v += hyp[b * 32 + u] * Wb[u * cout + co];
  bout[t] = v;
}

// ---------------------------------------------------------------------------
// Input prep: NHWC fp32 (B,256,256,2) -> NHWC bf16 (B,256,256,32) zero-padded
// ---------------------------------------------------------------------------
__global__ void prep_kernel(const float* __restrict__ zf, bf16_t* __restrict__ out, int HW) {
  int idx = blockIdx.x * blockDim.x + threadIdx.x; // b*HW + p
  if (idx >= 4 * HW) return;
  v8bf z;
#pragma unroll
  for (int i = 0; i < 8; ++i) z[i] = (bf16_t)0.f;
  v8bf first = z;
  first[0] = (bf16_t)zf[(size_t)idx * 2 + 0];
  first[1] = (bf16_t)zf[(size_t)idx * 2 + 1];
  bf16_t* o = out + (size_t)idx * 32;
  *(v8bf*)(o + 0)  = first;
  *(v8bf*)(o + 8)  = z;
  *(v8bf*)(o + 16) = z;
  *(v8bf*)(o + 24) = z;
}

// ---------------------------------------------------------------------------
// Implicit-GEMM 3x3 conv (pad 1), cout=64, cin in {32,64,128}, via WMMA bf16.
// Per-sample weights w: [B][9][64][cin] bf16 (TDM-staged), bias [B][64] fp32.
// Activations NHWC bf16 (async-staged). Optional concat second source.
// WG = 256 thr (8 waves). Tile = 16 rows x 16 cols; wave w -> rows w and w+8,
// 8 accumulators; A-fragments reused across the two row-tiles.
// ---------------------------------------------------------------------------
__device__ __forceinline__ v16bf frag_load(const bf16_t* p, int second_off) {
  v8bf lo = *(const v8bf*)(p);
  v8bf hi = *(const v8bf*)(p + second_off);
  v16bf r;
#pragma unroll
  for (int i = 0; i < 8; ++i) { r[i] = lo[i]; r[8 + i] = hi[i]; }
  return r;
}

__global__ __launch_bounds__(256)
void conv3x3_wmma_kernel(const bf16_t* __restrict__ in0, const bf16_t* __restrict__ in1,
                         const bf16_t* __restrict__ w, const float* __restrict__ bias,
                         bf16_t* __restrict__ out,
                         int H, int W, int cin, int cin0, int relu) {
  extern __shared__ char smem[];
  bf16_t* w_lds  = (bf16_t*)smem;
  bf16_t* in_lds = (bf16_t*)(smem + (size_t)9 * 64 * cin * sizeof(bf16_t));

  const int TW = 16, TH = 16;
  const int tid = threadIdx.x;
  const int x0 = blockIdx.x * TW;
  const int y0 = blockIdx.y * TH;
  const int b  = blockIdx.z;

  // --- stage per-sample weights [9][64][cin] into LDS -----------------------
#if HAVE_TDM
  {
    const int wave = tid >> 5;                         // wave-uniform
    const unsigned n8_total = (unsigned)(9 * 64 * cin) / 4;  // 8B elems
    const unsigned n8w = n8_total >> 3;                // slice per wave
    unsigned lds_off = (unsigned)(unsigned long long)(lds_char_t*)(char*)w_lds
                     + (unsigned)wave * n8w * 8u;
    const bf16_t* gsrc = w + (size_t)b * 9 * 64 * cin + (size_t)wave * n8w * 4u;
    tdm_copy_1d(gsrc, lds_off, n8w);
  }
#else
  {
    const int4* src = (const int4*)(w + (size_t)b * 9 * 64 * cin);
    int4* dst = (int4*)w_lds;
    const int n16 = 9 * 64 * cin / 8;
    for (int i = tid; i < n16; i += 256) dst[i] = src[i];
  }
#endif

  // --- stage haloed input tile (TH+2)x(TW+2)xcin, zero halo -----------------
  {
    const int cg = cin >> 3;
    const int n16 = (TH + 2) * (TW + 2) * cg;
    for (int i = tid; i < n16; i += 256) {
      int c8  = (i % cg) << 3;
      int pos = i / cg;
      int ty = pos / (TW + 2), tx = pos % (TW + 2);
      int gy = y0 - 1 + ty, gx = x0 - 1 + tx;
      bf16_t* dst = in_lds + (size_t)pos * cin + c8;
      if ((unsigned)gy < (unsigned)H && (unsigned)gx < (unsigned)W) {
        size_t p = (size_t)((b * H + gy) * W + gx);
        const bf16_t* src = (c8 < cin0) ? (in0 + p * cin0 + c8)
                                        : (in1 + p * (cin - cin0) + (c8 - cin0));
#if HAVE_ASYNC_COPY
        __builtin_amdgcn_global_load_async_to_lds_b128(
            (g_v4i*)(const void*)src, (l_v4i*)(void*)dst, 0, 0);
#else
        *(int4*)dst = *(const int4*)src;
#endif
      } else {
        *(int4*)dst = make_int4(0, 0, 0, 0);
      }
    }
  }

  // --- drain async/tensor counters, then block barrier ----------------------
#if HAVE_TDM
  __builtin_amdgcn_s_wait_tensorcnt(0);
#endif
#if HAVE_ASYNC_COPY
#if __has_builtin(__builtin_amdgcn_s_wait_asynccnt)
  __builtin_amdgcn_s_wait_asynccnt(0);
#else
  asm volatile("s_wait_asynccnt 0x0" ::: "memory");
#endif
#endif
  __syncthreads();

  const int lane = tid & 31;
  const int r    = tid >> 5;        // wave id: output rows r and r+8
  const int lN   = lane & 15;
  const int hi   = lane >> 4;       // lane half
  const int aoff = hi * 8;          // A: K-run start {0,8}, second run +16
  const int boff = hi * 8;          // B: K-run start {0,8}, second run +16

  v8f acc0[4], acc1[4];
#pragma unroll
  for (int m = 0; m < 4; ++m)
#pragma unroll
    for (int i = 0; i < 8; ++i) { acc0[m][i] = 0.f; acc1[m][i] = 0.f; }

  for (int tap = 0; tap < 9; ++tap) {
    const int dy = tap / 3, dx = tap % 3;
    const bf16_t* bbase0 = in_lds + ((size_t)((r + dy) * (TW + 2) + lN + dx)) * cin;
    const bf16_t* bbase1 = in_lds + ((size_t)((r + 8 + dy) * (TW + 2) + lN + dx)) * cin;
    const bf16_t* abase  = w_lds + ((size_t)(tap * 64 + lN)) * cin;
    for (int kk = 0; kk < cin; kk += 32) {
      v16bf bf0 = frag_load(bbase0 + kk + boff, 16);
      v16bf bf1 = frag_load(bbase1 + kk + boff, 16);
#pragma unroll
      for (int m = 0; m < 4; ++m) {
        v16bf afrag = frag_load(abase + (size_t)m * 16 * cin + kk + aoff, 16);
        acc0[m] = __builtin_amdgcn_wmma_f32_16x16x32_bf16(
            false, afrag, false, bf0, (short)0, acc0[m], false, false);
        acc1[m] = __builtin_amdgcn_wmma_f32_16x16x32_bf16(
            false, afrag, false, bf1, (short)0, acc1[m], false, false);
      }
    }
  }

  // epilogue: C/D layout -> lane (N=lN), vgpr v -> M = v + 8*hi; co = m*16+ms+v
  const int ms = hi * 8;
  const int gx = x0 + lN;
  bf16_t* obase0 = out + ((size_t)((b * H + y0 + r)     * W + gx)) * 64;
  bf16_t* obase1 = out + ((size_t)((b * H + y0 + r + 8) * W + gx)) * 64;
#pragma unroll
  for (int m = 0; m < 4; ++m) {
    const int co0 = m * 16 + ms;
    v8bf o0, o1;
#pragma unroll
    for (int v = 0; v < 8; ++v) {
      float bv = bias[b * 64 + co0 + v];
      float x0v = acc0[m][v] + bv;
      float x1v = acc1[m][v] + bv;
      if (relu) { x0v = x0v > 0.f ? x0v : 0.f; x1v = x1v > 0.f ? x1v : 0.f; }
      o0[v] = (bf16_t)x0v;
      o1[v] = (bf16_t)x1v;
    }
    *(v8bf*)(obase0 + co0) = o0;
    *(v8bf*)(obase1 + co0) = o1;
  }
}

// ---------------------------------------------------------------------------
// 2x2 maxpool, NHWC bf16, 64 channels; thread = 8 channels of one out pixel
// ---------------------------------------------------------------------------
__global__ void maxpool_kernel(const bf16_t* __restrict__ in, bf16_t* __restrict__ out,
                               int H, int W, int total8) {
  int idx = blockIdx.x * blockDim.x + threadIdx.x;
  if (idx >= total8) return;
  int cg = (idx & 7) << 3;
  int t = idx >> 3;
  int Wo = W >> 1, Ho = H >> 1;
  int x = t % Wo; t /= Wo;
  int y = t % Ho; int b = t / Ho;
  size_t base = ((size_t)(b * H + y * 2) * W + x * 2) * 64 + cg;
  v8bf p0 = *(const v8bf*)(in + base);
  v8bf p1 = *(const v8bf*)(in + base + 64);
  v8bf p2 = *(const v8bf*)(in + base + (size_t)W * 64);
  v8bf p3 = *(const v8bf*)(in + base + (size_t)W * 64 + 64);
  v8bf o;
#pragma unroll
  for (int i = 0; i < 8; ++i) {
    float m = fmaxf(fmaxf((float)p0[i], (float)p1[i]), fmaxf((float)p2[i], (float)p3[i]));
    o[i] = (bf16_t)m;
  }
  *(v8bf*)(out + ((size_t)(b * Ho + y) * Wo + x) * 64 + cg) = o;
}

// ---------------------------------------------------------------------------
// 2x bilinear upsample, align_corners=True semantics, NHWC bf16, 64 channels
// ---------------------------------------------------------------------------
__global__ void upsample_kernel(const bf16_t* __restrict__ in, bf16_t* __restrict__ out,
                                int H, int W, int total8) {
  int idx = blockIdx.x * blockDim.x + threadIdx.x;
  if (idx >= total8) return;
  int cg = (idx & 7) << 3;
  int t = idx >> 3;
  int Wo = 2 * W, Ho = 2 * H;
  int x = t % Wo; t /= Wo;
  int y = t % Ho; int b = t / Ho;

  float py = (float)y * (float)(H - 1) / (float)(2 * H - 1);
  int ly = (int)floorf(py); ly = ly < 0 ? 0 : (ly > H - 2 ? H - 2 : ly);
  float fy = py - (float)ly;
  float px = (float)x * (float)(W - 1) / (float)(2 * W - 1);
  int lx = (int)floorf(px); lx = lx < 0 ? 0 : (lx > W - 2 ? W - 2 : lx);
  float fx = px - (float)lx;

  size_t base = ((size_t)(b * H + ly) * W + lx) * 64 + cg;
  v8bf p00 = *(const v8bf*)(in + base);
  v8bf p01 = *(const v8bf*)(in + base + 64);
  v8bf p10 = *(const v8bf*)(in + base + (size_t)W * 64);
  v8bf p11 = *(const v8bf*)(in + base + (size_t)W * 64 + 64);
  v8bf o;
#pragma unroll
  for (int i = 0; i < 8; ++i) {
    float a = (float)p00[i] * (1.f - fy) + (float)p10[i] * fy;
    float c = (float)p01[i] * (1.f - fy) + (float)p11[i] * fy;
    o[i] = (bf16_t)(a * (1.f - fx) + c * fx);
  }
  *(v8bf*)(out + ((size_t)(b * Ho + y) * Wo + x) * 64 + cg) = o;
}

// ---------------------------------------------------------------------------
// Final layer: 1x1 conv 64->2 (per-sample weights [B][2][64]) + residual zf
// ---------------------------------------------------------------------------
__global__ void final_kernel(const float* __restrict__ zf, const bf16_t* __restrict__ act,
                             const bf16_t* __restrict__ w, const float* __restrict__ bias,
                             float* __restrict__ out, int HW) {
  int idx = blockIdx.x * blockDim.x + threadIdx.x; // b*HW + p
  if (idx >= 4 * HW) return;
  int b = idx / HW;
  const bf16_t* a = act + (size_t)idx * 64;
#pragma unroll
  for (int co = 0; co < 2; ++co) {
    const bf16_t* ww = w + ((size_t)b * 2 + co) * 64;
    float s = bias[b * 2 + co];
    for (int i = 0; i < 64; ++i) s += (float)a[i] * (float)ww[i];
    out[(size_t)idx * 2 + co] = zf[(size_t)idx * 2 + co] + s;
  }
}

// ---------------------------------------------------------------------------
extern "C" void kernel_launch(void* const* d_in, const int* in_sizes, int n_in,
                              void* d_out, int out_size, void* d_ws, size_t ws_size,
                              hipStream_t stream) {
  const int B = 4, H0 = 256;
  const float* zf  = (const float*)d_in[0];
  const float* hp  = (const float*)d_in[2];
  const float* hW1 = (const float*)d_in[3];
  const float* hb1 = (const float*)d_in[4];
  const float* hW2 = (const float*)d_in[5];
  const float* hb2 = (const float*)d_in[6];
  const float* hWo = (const float*)d_in[7];
  const float* hbo = (const float*)d_in[8];
  auto LP = [&](int i, int j) { return (const float*)d_in[9 + 4 * i + j]; };

  char* ws = (char*)d_ws;
  size_t off = 0;
  auto alloc = [&](size_t bytes) -> void* {
    size_t cur = (off + 255) & ~(size_t)255;
    off = cur + bytes;
    return (void*)(ws + cur);
  };

  float* hyp = (float*)alloc((size_t)B * 32 * sizeof(float));

  static const int cinA[15]  = {2,64,64,64,64,64,64,64,128,64,128,64,128,64,64};
  static const int cinP[15]  = {32,64,64,64,64,64,64,64,128,64,128,64,128,64,64};
  static const int coutA[15] = {64,64,64,64,64,64,64,64,64,64,64,64,64,64,2};
  static const int ksA[15]   = {3,3,3,3,3,3,3,3,3,3,3,3,3,3,1};

  bf16_t* wb[15]; float* bb[15];
  for (int i = 0; i < 15; ++i) {
    wb[i] = (bf16_t*)alloc((size_t)B * ksA[i] * ksA[i] * coutA[i] * cinP[i] * sizeof(bf16_t));
    bb[i] = (float*)alloc((size_t)B * coutA[i] * sizeof(float));
  }

  auto abuf = [&](int HW, int C) { return (bf16_t*)alloc((size_t)B * HW * HW * C * sizeof(bf16_t)); };
  bf16_t* act0  = abuf(256, 32);
  bf16_t* a256a = abuf(256, 64);
  bf16_t* skip1 = abuf(256, 64);
  bf16_t* a256b = abuf(256, 64);
  bf16_t* up256 = abuf(256, 64);
  bf16_t* pool1 = abuf(128, 64);
  bf16_t* t128a = abuf(128, 64);
  bf16_t* skip2 = abuf(128, 64);
  bf16_t* t128b = abuf(128, 64);
  bf16_t* up128 = abuf(128, 64);
  bf16_t* pool2 = abuf(64, 64);
  bf16_t* t64a  = abuf(64, 64);
  bf16_t* skip3 = abuf(64, 64);
  bf16_t* t64b  = abuf(64, 64);
  bf16_t* up64  = abuf(64, 64);
  bf16_t* pool3 = abuf(32, 64);
  bf16_t* t32a  = abuf(32, 64);
  bf16_t* convS = abuf(32, 64);

  hypernet_kernel<<<1, 128, 0, stream>>>(hp, hW1, hb1, hW2, hb2, hWo, hbo, hyp);

  for (int i = 0; i < 15; ++i) {
    int total = B * ksA[i] * ksA[i] * coutA[i] * cinP[i];
    genw_kernel<<<(total + 255) / 256, 256, 0, stream>>>(
        hyp, LP(i, 0), LP(i, 1), wb[i], cinA[i], cinP[i], coutA[i], ksA[i], total);
    genb_kernel<<<1, 256, 0, stream>>>(hyp, LP(i, 2), LP(i, 3), bb[i], coutA[i]);
  }

  prep_kernel<<<(B * H0 * H0 + 255) / 256, 256, 0, stream>>>(zf, act0, H0 * H0);

  auto conv = [&](int i, const bf16_t* in0p, const bf16_t* in1p, bf16_t* outp, int H) {
    int cin = cinP[i];
    int cin0 = in1p ? 64 : cin;
    dim3 grid(H / 16, H / 16, B);
    size_t sh = ((size_t)9 * 64 * cin + (size_t)18 * 18 * cin) * sizeof(bf16_t);
    conv3x3_wmma_kernel<<<grid, 256, sh, stream>>>(in0p, in1p, wb[i], bb[i], outp,
                                                   H, H, cin, cin0, 1);
  };
  auto pool = [&](const bf16_t* inp, bf16_t* outp, int H) {
    int total8 = B * (H / 2) * (H / 2) * 8;
    maxpool_kernel<<<(total8 + 255) / 256, 256, 0, stream>>>(inp, outp, H, H, total8);
  };
  auto ups = [&](const bf16_t* inp, bf16_t* outp, int H) {
    int total8 = B * (2 * H) * (2 * H) * 8;
    upsample_kernel<<<(total8 + 255) / 256, 256, 0, stream>>>(inp, outp, H, H, total8);
  };

  conv(0, act0, nullptr, a256a, 256);
  conv(1, a256a, nullptr, skip1, 256);
  pool(skip1, pool1, 256);
  conv(2, pool1, nullptr, t128a, 128);
  conv(3, t128a, nullptr, skip2, 128);
  pool(skip2, pool2, 128);
  conv(4, pool2, nullptr, t64a, 64);
  conv(5, t64a, nullptr, skip3, 64);
  pool(skip3, pool3, 64);
  conv(6, pool3, nullptr, t32a, 32);
  conv(7, t32a, nullptr, convS, 32);
  ups(convS, up64, 32);
  conv(8, up64, skip3, t64a, 64);
  conv(9, t64a, nullptr, t64b, 64);
  ups(t64b, up128, 64);
  conv(10, up128, skip2, t128a, 128);
  conv(11, t128a, nullptr, t128b, 128);
  ups(t128b, up256, 128);
  conv(12, up256, skip1, a256a, 256);
  conv(13, a256a, nullptr, a256b, 256);

  final_kernel<<<(B * H0 * H0 + 255) / 256, 256, 0, stream>>>(
      zf, a256b, wb[14], bb[14], (float*)d_out, H0 * H0);

  (void)in_sizes; (void)n_in; (void)out_size; (void)ws_size;
}
